// ObjectIdentifier_77429670412475
// MI455X (gfx1250) — matrix operand order repaired and checked by reading, not compiled, not run on hardware
//
#include <hip/hip_runtime.h>
#include <hip/hip_bf16.h>
#include <stdint.h>

// Flip to 0 if the async-to-LDS inline asm fails to assemble on this toolchain.
#define ASYNC_STAGING 1

// ---------------------------------------------------------------------------
// Types for CDNA5 WMMA (wave32): v_wmma_f32_16x16x32_bf16
// ---------------------------------------------------------------------------
typedef __bf16 bf16_t;
typedef __attribute__((ext_vector_type(16))) __bf16 v16bf;
typedef __attribute__((ext_vector_type(8)))  __bf16 bf16x8;
typedef __attribute__((ext_vector_type(8)))  float  v8f;

__device__ __forceinline__ bf16_t f2bf(float f) {
  // round-to-nearest-even f32 -> bf16 (bit-level)
  unsigned int u = __builtin_bit_cast(unsigned int, f);
  unsigned int lsb = (u >> 16) & 1u;
  u += 0x7fffu + lsb;
  unsigned short h = (unsigned short)(u >> 16);
  return __builtin_bit_cast(bf16_t, h);
}

// Generic (shared) pointer -> 32-bit LDS byte offset for async-to-LDS ops.
__device__ __forceinline__ unsigned lds_off(const void* p) {
  return (unsigned)(unsigned long long)(__attribute__((address_space(3))) const void*)p;
}

#if ASYNC_STAGING
// GLOBAL_LOAD_ASYNC_TO_LDS_B128: per-lane 16B global->LDS copy, tracked by ASYNCcnt.
__device__ __forceinline__ void async_copy_b128(const void* gaddr, unsigned laddr) {
  asm volatile("global_load_async_to_lds_b128 %0, %1, off"
               :: "v"(laddr), "v"(gaddr) : "memory");
}
#if __has_builtin(__builtin_amdgcn_s_wait_asynccnt)
#define WAIT_ASYNC(n) __builtin_amdgcn_s_wait_asynccnt(n)
#else
#define WAIT_ASYNC(n) asm volatile("s_wait_asynccnt %0" :: "n"(n) : "memory")
#endif
#endif

// ---------------------------------------------------------------------------
// Problem sizes (fixed by the reference)
// ---------------------------------------------------------------------------
#define BATCH 8192
#define DIMG  768
#define DCAT  1536
#define N1    1024
#define N2    512
#define N3    256
#define NEMB  10000

// ---------------------------------------------------------------------------
// Prep kernels
// ---------------------------------------------------------------------------
__global__ __launch_bounds__(256) void convert_concat_x(
    const float* __restrict__ img, const float* __restrict__ txt,
    bf16_t* __restrict__ x) {
  size_t idx = (size_t)blockIdx.x * 256 + threadIdx.x;
  if (idx >= (size_t)BATCH * DCAT) return;
  int m = (int)(idx / DCAT), c = (int)(idx % DCAT);
  float v = (c < DIMG) ? img[(size_t)m * DIMG + c] : txt[(size_t)m * DIMG + (c - DIMG)];
  x[idx] = f2bf(v);
}

// Wt[n][k] = bf16(W[k][n])   (W is [K][N] row-major in memory)
__global__ __launch_bounds__(256) void transpose_convert_w(
    const float* __restrict__ W, bf16_t* __restrict__ Wt, int K, int N) {
  size_t idx = (size_t)blockIdx.x * 256 + threadIdx.x;
  if (idx >= (size_t)K * N) return;
  int n = (int)(idx / K), k = (int)(idx % K);
  Wt[idx] = f2bf(W[(size_t)k * N + n]);
}

// For a [R][256] f32 matrix: per-row L2 norm (f32) + bf16 copy. One wave per row.
__global__ __launch_bounds__(256) void norm_convert_256(
    const float* __restrict__ X, bf16_t* __restrict__ Xb,
    float* __restrict__ norms, int R) {
  int w = threadIdx.x >> 5, lane = threadIdx.x & 31;
  int r = blockIdx.x * 8 + w;
  if (r >= R) return;
  const float* row = X + (size_t)r * 256;
  float vals[8]; float s = 0.f;
#pragma unroll
  for (int e = 0; e < 8; ++e) { float f = row[lane * 8 + e]; vals[e] = f; s += f * f; }
#pragma unroll
  for (int off = 16; off; off >>= 1) s += __shfl_xor(s, off, 32);
  if (lane == 0) norms[r] = sqrtf(s);
#pragma unroll
  for (int e = 0; e < 8; ++e) Xb[(size_t)r * 256 + lane * 8 + e] = f2bf(vals[e]);
}

// ---------------------------------------------------------------------------
// WMMA GEMM:  Out[M][N] = epilogue( A[M][K](bf16) x Bt[N][K](bf16)^T )
//  MODE 0: relu(acc + bias[n])            -> bf16 out
//  MODE 1: acc + bias[n]                  -> f32 out
//  MODE 2: acc / max(rown[m]*coln[n],eps) -> f32 out (n guarded against N)
// Block tile 128x128, 8 waves (4x2), wave tile 32x64 = 2x4 WMMA tiles, KC=32.
// Staging: LDS double-buffer fed by GLOBAL_LOAD_ASYNC_TO_LDS_B128 (ASYNCcnt),
// 4 async b128 per thread per stage; s_wait_asynccnt 4 leaves only the
// just-issued next stage outstanding. Fragment layouts per ISA 7.12.2.
// ---------------------------------------------------------------------------
template <int MODE>
__global__ __launch_bounds__(256) void gemm_wmma_bf16(
    const bf16_t* __restrict__ A, const bf16_t* __restrict__ Bt,
    const float* __restrict__ bias,
    const float* __restrict__ rown, const float* __restrict__ coln,
    void* __restrict__ Out, int M, int N, int K) {
#if ASYNC_STAGING
  __shared__ bf16_t As[2][128][40];  // 32 live cols + 8 pad (80B rows, 16B aligned)
  __shared__ bf16_t Bs[2][128][40];
#else
  __shared__ bf16_t As[1][128][40];
  __shared__ bf16_t Bs[1][128][40];
#endif

  const int tid = threadIdx.x;
  const int bm = blockIdx.y, bn = blockIdx.x;
  const int wid = tid >> 5, lane = tid & 31;
  const int wm = wid >> 1, wn = wid & 1;        // wave grid 4x2
  const int half = lane >> 4, l15 = lane & 15;

  v8f acc[2][4];
#pragma unroll
  for (int i = 0; i < 2; ++i)
#pragma unroll
    for (int j = 0; j < 4; ++j) acc[i][j] = (v8f)0.f;

  const int KT = K / 32;

#if ASYNC_STAGING
  auto issueStage = [&](int kt, int buf) {
#pragma unroll
    for (int i = 0; i < 2; ++i) {
      int v = tid + i * 256, row = v >> 2, c8 = v & 3;
      const bf16_t* ga = A + (size_t)(bm * 128 + row) * K + kt * 32 + c8 * 8;
      async_copy_b128(ga, lds_off(&As[buf][row][c8 * 8]));
      int n = bn * 128 + row;
      n = (n < N) ? n : (N - 1);  // clamp: ragged cols are discarded in epilogue
      const bf16_t* gb = Bt + (size_t)n * K + kt * 32 + c8 * 8;
      async_copy_b128(gb, lds_off(&Bs[buf][row][c8 * 8]));
    }
  };
  issueStage(0, 0);
#else
  uint4 ra[2], rb[2];
  auto loadA = [&](int kt) {
#pragma unroll
    for (int i = 0; i < 2; ++i) {
      int v = tid + i * 256, row = v >> 2, c8 = v & 3;
      ra[i] = *(const uint4*)(A + (size_t)(bm * 128 + row) * K + kt * 32 + c8 * 8);
    }
  };
  auto loadB = [&](int kt) {
#pragma unroll
    for (int i = 0; i < 2; ++i) {
      int v = tid + i * 256, row = v >> 2, c8 = v & 3;
      int n = bn * 128 + row;
      rb[i] = (n < N) ? *(const uint4*)(Bt + (size_t)n * K + kt * 32 + c8 * 8)
                      : make_uint4(0u, 0u, 0u, 0u);
    }
  };
  auto storeLDS = [&]() {
#pragma unroll
    for (int i = 0; i < 2; ++i) {
      int v = tid + i * 256, row = v >> 2, c8 = v & 3;
      *(uint4*)&As[0][row][c8 * 8] = ra[i];
      *(uint4*)&Bs[0][row][c8 * 8] = rb[i];
    }
  };
  loadA(0); loadB(0);
  storeLDS();
#endif

  for (int kt = 0; kt < KT; ++kt) {
#if ASYNC_STAGING
    const int cur = kt & 1;
    if (kt + 1 < KT) {
      issueStage(kt + 1, cur ^ 1);  // prev buffer free: end-of-iter barrier passed
      WAIT_ASYNC(4);                // stage kt's 4 copies done; kt+1 in flight
    } else {
      WAIT_ASYNC(0);
    }
    __syncthreads();
#else
    const int cur = 0;
    __syncthreads();
    if (kt + 1 < KT) { loadA(kt + 1); loadB(kt + 1); }
#endif

    // A fragments: lane holds row M=l15; K runs at 8*half and 16+8*half
    v16bf af[2];
#pragma unroll
    for (int i = 0; i < 2; ++i) {
      int r = wm * 32 + i * 16 + l15;
      bf16x8 lo = *(const bf16x8*)&As[cur][r][half * 8];
      bf16x8 hi = *(const bf16x8*)&As[cur][r][16 + half * 8];
#pragma unroll
      for (int e = 0; e < 8; ++e) { af[i][e] = lo[e]; af[i][e + 8] = hi[e]; }
    }
    // B fragments: lane holds col N=l15; K runs at 8*half and 16+8*half
    v16bf bfg[4];
#pragma unroll
    for (int j = 0; j < 4; ++j) {
      int r = wn * 64 + j * 16 + l15;
      bf16x8 lo = *(const bf16x8*)&Bs[cur][r][half * 8];
      bf16x8 hi = *(const bf16x8*)&Bs[cur][r][16 + half * 8];
#pragma unroll
      for (int e = 0; e < 8; ++e) { bfg[j][e] = lo[e]; bfg[j][e + 8] = hi[e]; }
    }

#pragma unroll
    for (int i = 0; i < 2; ++i)
#pragma unroll
      for (int j = 0; j < 4; ++j)
        acc[i][j] = __builtin_amdgcn_wmma_f32_16x16x32_bf16(
            false, af[i], false, bfg[j], (short)0, acc[i][j], false, false);

    __syncthreads();
#if !ASYNC_STAGING
    if (kt + 1 < KT) storeLDS();
#endif
  }

  // Epilogue. C layout: elem v -> M = base + 8*half + v, N = base + l15.
#pragma unroll
  for (int i = 0; i < 2; ++i) {
    int mb = bm * 128 + wm * 32 + i * 16 + half * 8;
#pragma unroll
    for (int j = 0; j < 4; ++j) {
      int n = bn * 128 + wn * 64 + j * 16 + l15;
      if (MODE == 2 && n >= N) continue;
      float bv = (MODE == 2) ? 0.f : bias[n];
      float env = (MODE == 2) ? coln[n] : 0.f;
#pragma unroll
      for (int v = 0; v < 8; ++v) {
        int m = mb + v;
        float c = acc[i][j][v];
        if (MODE == 0) {
          float r = c + bv; r = r > 0.f ? r : 0.f;
          ((bf16_t*)Out)[(size_t)m * N + n] = f2bf(r);
        } else if (MODE == 1) {
          ((float*)Out)[(size_t)m * N + n] = c + bv;
        } else {
          float d = rown[m] * env;
          d = d > 1e-8f ? d : 1e-8f;
          ((float*)Out)[(size_t)m * N + n] = c / d;
        }
      }
    }
  }
}

// ---------------------------------------------------------------------------
// Top-5 per row (one 256-thread block per row); indices written as float
// ---------------------------------------------------------------------------
__global__ __launch_bounds__(256) void topk5_rows(
    const float* __restrict__ sims, float* __restrict__ outIdx, int N) {
  const int row = blockIdx.x, tid = threadIdx.x;
  float v[5]; int id[5];
#pragma unroll
  for (int k = 0; k < 5; ++k) { v[k] = -__builtin_inff(); id[k] = 0x7fffffff; }
  const float* s = sims + (size_t)row * N;
  for (int c = tid; c < N; c += 256) {
    float val = s[c];
    if (val > v[4]) {
      v[4] = val; id[4] = c;
#pragma unroll
      for (int k = 4; k > 0; --k) {
        if (v[k] > v[k - 1] || (v[k] == v[k - 1] && id[k] < id[k - 1])) {
          float tv = v[k]; v[k] = v[k - 1]; v[k - 1] = tv;
          int ti = id[k]; id[k] = id[k - 1]; id[k - 1] = ti;
        }
      }
    }
  }
  __shared__ float sv[256][5];
  __shared__ int   si[256][5];
#pragma unroll
  for (int k = 0; k < 5; ++k) { sv[tid][k] = v[k]; si[tid][k] = id[k]; }
  __syncthreads();
  for (int stride = 128; stride >= 1; stride >>= 1) {
    if (tid < stride) {
      float a0[5], b0[5]; int ia[5], ib[5];
#pragma unroll
      for (int k = 0; k < 5; ++k) {
        a0[k] = sv[tid][k];          ia[k] = si[tid][k];
        b0[k] = sv[tid + stride][k]; ib[k] = si[tid + stride][k];
      }
      float mv[5]; int mi[5]; int pa = 0, pb = 0;
#pragma unroll
      for (int k = 0; k < 5; ++k) {
        bool takeA = (a0[pa] > b0[pb]) || (a0[pa] == b0[pb] && ia[pa] <= ib[pb]);
        if (takeA) { mv[k] = a0[pa]; mi[k] = ia[pa]; ++pa; }
        else       { mv[k] = b0[pb]; mi[k] = ib[pb]; ++pb; }
      }
#pragma unroll
      for (int k = 0; k < 5; ++k) { sv[tid][k] = mv[k]; si[tid][k] = mi[k]; }
    }
    __syncthreads();
  }
  if (tid == 0) {
#pragma unroll
    for (int k = 0; k < 5; ++k) outIdx[(size_t)row * 5 + k] = (float)si[0][k];
  }
}

// ---------------------------------------------------------------------------
// Host-side orchestration
// ---------------------------------------------------------------------------
extern "C" void kernel_launch(void* const* d_in, const int* in_sizes, int n_in,
                              void* d_out, int out_size, void* d_ws, size_t ws_size,
                              hipStream_t stream) {
  const float* img = (const float*)d_in[0];
  const float* txt = (const float*)d_in[1];
  const float* W1  = (const float*)d_in[2];
  const float* b1  = (const float*)d_in[3];
  const float* W2  = (const float*)d_in[4];
  const float* b2  = (const float*)d_in[5];
  const float* W3  = (const float*)d_in[6];
  const float* b3  = (const float*)d_in[7];
  const float* emb = (const float*)d_in[8];
  float* out = (float*)d_out;

  char* ws = (char*)d_ws;
  size_t off = 0;
  auto alloc = [&](size_t bytes) -> char* {
    char* p = ws + off;
    off += (bytes + 255) & ~(size_t)255;
    return p;
  };
  bf16_t* xb   = (bf16_t*)alloc((size_t)BATCH * DCAT * 2);
  bf16_t* w1t  = (bf16_t*)alloc((size_t)N1 * DCAT * 2);
  bf16_t* w2t  = (bf16_t*)alloc((size_t)N2 * N1 * 2);
  bf16_t* w3t  = (bf16_t*)alloc((size_t)N3 * N2 * 2);
  bf16_t* embb = (bf16_t*)alloc((size_t)NEMB * N3 * 2);
  bf16_t* h1   = (bf16_t*)alloc((size_t)BATCH * N1 * 2);
  bf16_t* h2   = (bf16_t*)alloc((size_t)BATCH * N2 * 2);
  float*  pf   = (float*) alloc((size_t)BATCH * N3 * 4);
  bf16_t* pb   = (bf16_t*)alloc((size_t)BATCH * N3 * 2);
  float*  pn   = (float*) alloc((size_t)BATCH * 4);
  float*  en   = (float*) alloc((size_t)NEMB * 4);

  // 1) concat + bf16 convert of x
  convert_concat_x<<<(unsigned)(((size_t)BATCH * DCAT + 255) / 256), 256, 0, stream>>>(img, txt, xb);
  // 2) transposed bf16 weights ([N][K] so GEMM B-tiles are K-contiguous)
  transpose_convert_w<<<(unsigned)(((size_t)DCAT * N1 + 255) / 256), 256, 0, stream>>>(W1, w1t, DCAT, N1);
  transpose_convert_w<<<(unsigned)(((size_t)N1 * N2 + 255) / 256), 256, 0, stream>>>(W2, w2t, N1, N2);
  transpose_convert_w<<<(unsigned)(((size_t)N2 * N3 + 255) / 256), 256, 0, stream>>>(W3, w3t, N2, N3);
  // 3) emb norms + bf16 emb (emb is already [N][K])
  norm_convert_256<<<(NEMB + 7) / 8, 256, 0, stream>>>(emb, embb, en, NEMB);

  // 4) MLP via WMMA
  gemm_wmma_bf16<0><<<dim3(N1 / 128, BATCH / 128), 256, 0, stream>>>(
      xb, w1t, b1, nullptr, nullptr, h1, BATCH, N1, DCAT);
  gemm_wmma_bf16<0><<<dim3(N2 / 128, BATCH / 128), 256, 0, stream>>>(
      h1, w2t, b2, nullptr, nullptr, h2, BATCH, N2, N1);
  gemm_wmma_bf16<1><<<dim3(N3 / 128, BATCH / 128), 256, 0, stream>>>(
      h2, w3t, b3, nullptr, nullptr, pf, BATCH, N3, N2);

  // 5) p norms + bf16 p
  norm_convert_256<<<BATCH / 8, 256, 0, stream>>>(pf, pb, pn, BATCH);

  // 6) cosine sims via WMMA with fused 1/max(pn*en,eps) epilogue -> d_out
  gemm_wmma_bf16<2><<<dim3((NEMB + 127) / 128, BATCH / 128), 256, 0, stream>>>(
      pb, embb, nullptr, pn, en, out, BATCH, NEMB, N3);

  // 7) top-5 indices (as float) appended after sims
  topk5_rows<<<BATCH, 256, 0, stream>>>(out, out + (size_t)BATCH * NEMB, NEMB);

  (void)in_sizes; (void)n_in; (void)out_size; (void)ws_size;
}
